// GA_61040075210975
// MI455X (gfx1250) — compile-verified, matrix operand order checked by reading, not run on hardware
//
#include <hip/hip_runtime.h>
#include <math.h>

// ---------------------------------------------------------------------------
// Types for CDNA5 WMMA (wave32)
// ---------------------------------------------------------------------------
typedef __attribute__((ext_vector_type(16))) __bf16 bf16x16;
typedef __attribute__((ext_vector_type(8)))  float  f32x8;

union FragB {
    bf16x16 v;
    uint4   q[2];
};

__device__ __forceinline__ unsigned short f2bf(float f) {
    union { float f; unsigned int u; } c; c.f = f;
    unsigned int u = c.u;
    unsigned int r = u + 0x7FFFu + ((u >> 16) & 1u);   // round-to-nearest-even
    return (unsigned short)(r >> 16);
}

// sigmoid(x*0.125) = 1 / (1 + exp2(x * (-0.125*log2(e))))
#define SIGK (-0.125f * 1.44269504088896340736f)

// Problem constants
#define BB   4
#define LQ   1024
#define LK   2048
#define HH   1024
#define NHD  16
#define DHD  64
#define FFD  2048
#define MX   (BB * LQ)   // 4096
#define MY   (BB * LK)   // 8192

// ---------------------------------------------------------------------------
// fp32 -> bf16 elementwise convert (4 elements / thread)
// ---------------------------------------------------------------------------
__global__ __launch_bounds__(256) void cvt_bf16_kernel(
    const float* __restrict__ in, unsigned short* __restrict__ out, int n)
{
    int i = (blockIdx.x * 256 + threadIdx.x) * 4;
    if (i < n) {
        float4 v = *(const float4*)(in + i);
        out[i + 0] = f2bf(v.x);
        out[i + 1] = f2bf(v.y);
        out[i + 2] = f2bf(v.z);
        out[i + 3] = f2bf(v.w);
    }
}

// ---------------------------------------------------------------------------
// fp32 RxC -> bf16 CxR transpose-convert (weights -> N-major for WMMA B op)
// ---------------------------------------------------------------------------
__global__ __launch_bounds__(256) void transpose_cvt_kernel(
    const float* __restrict__ W, unsigned short* __restrict__ Wt, int R, int C)
{
    __shared__ float tile[32][33];
    int r0 = blockIdx.y * 32, c0 = blockIdx.x * 32;
    int tx = threadIdx.x & 31, ty = threadIdx.x >> 5;   // 32 x 8
#pragma unroll
    for (int i = 0; i < 32; i += 8)
        tile[ty + i][tx] = W[(size_t)(r0 + ty + i) * C + c0 + tx];
    __syncthreads();
#pragma unroll
    for (int i = 0; i < 32; i += 8)
        Wt[(size_t)(c0 + ty + i) * R + r0 + tx] = f2bf(tile[tx][ty + i]);
}

// ---------------------------------------------------------------------------
// Generic bf16 WMMA GEMM:  out = epilogue(A[MxK] * Bt[NxK]^T + bias)
// Block: 256 threads (8 waves), tile 128x128, BK=64.
// Wave (wr,wc) computes a 32x64 sub-tile = 2x4 WMMA 16x16 tiles.
// EPI: 0 = bias -> bf16 row-major
//      1 = bias -> bf16 scattered into V^T layout [B][NH][DH][Lk]
//      2 = bias + res(f32) -> f32 row-major
//      3 = relu(bias) -> bf16 row-major
// ---------------------------------------------------------------------------
template <int EPI>
__global__ __launch_bounds__(256) void gemm_bf16_kernel(
    const unsigned short* __restrict__ A,
    const unsigned short* __restrict__ Bt,
    const float* __restrict__ bias,
    const float* __restrict__ res,
    void* __restrict__ outp,
    int M, int N, int K, int LkV)
{
    __shared__ __align__(16) unsigned short As[128 * 72];
    __shared__ __align__(16) unsigned short Bs[128 * 72];

    const int tid  = threadIdx.x;
    const int lane = tid & 31;
    const int wid  = tid >> 5;
    const int hlf  = lane >> 4;      // 0: lanes 0-15, 1: lanes 16-31
    const int ln   = lane & 15;
    const int wr   = wid >> 1;       // 0..3
    const int wc   = wid & 1;        // 0..1
    const int m0   = blockIdx.y * 128;
    const int n0   = blockIdx.x * 128;

    f32x8 acc[2][4];
#pragma unroll
    for (int i = 0; i < 2; ++i)
#pragma unroll
        for (int j = 0; j < 4; ++j) acc[i][j] = f32x8{};

    const int ldRow = tid >> 1;          // 0..127
    const int ldSeg = (tid & 1) * 32;    // 0 or 32
    const unsigned short* gArow = A  + (size_t)(m0 + ldRow) * K + ldSeg;
    const unsigned short* gBrow = Bt + (size_t)(n0 + ldRow) * K + ldSeg;

    for (int k0 = 0; k0 < K; k0 += 64) {
        // prefetch next k-tile while this one is consumed (global_prefetch_b8)
        if (k0 + 64 < K) {
            __builtin_prefetch(gArow + k0 + 64, 0, 3);
            __builtin_prefetch(gBrow + k0 + 64, 0, 3);
        }
        const uint4* ga = (const uint4*)(gArow + k0);
        uint4* sa = (uint4*)&As[ldRow * 72 + ldSeg];
        sa[0] = ga[0]; sa[1] = ga[1]; sa[2] = ga[2]; sa[3] = ga[3];
        const uint4* gb = (const uint4*)(gBrow + k0);
        uint4* sb = (uint4*)&Bs[ldRow * 72 + ldSeg];
        sb[0] = gb[0]; sb[1] = gb[1]; sb[2] = gb[2]; sb[3] = gb[3];
        __syncthreads();

#pragma unroll
        for (int kc = 0; kc < 2; ++kc) {
            FragB af[2], bf[4];
#pragma unroll
            for (int i = 0; i < 2; ++i) {
                int r = wr * 32 + i * 16 + ln;
                af[i].q[0] = *(const uint4*)&As[r * 72 + kc * 32 + hlf * 8];
                af[i].q[1] = *(const uint4*)&As[r * 72 + kc * 32 + 16 + hlf * 8];
            }
#pragma unroll
            for (int j = 0; j < 4; ++j) {
                int r = wc * 64 + j * 16 + ln;
                bf[j].q[0] = *(const uint4*)&Bs[r * 72 + kc * 32 + hlf * 8];
                bf[j].q[1] = *(const uint4*)&Bs[r * 72 + kc * 32 + 16 + hlf * 8];
            }
#pragma unroll
            for (int i = 0; i < 2; ++i)
#pragma unroll
                for (int j = 0; j < 4; ++j)
                    acc[i][j] = __builtin_amdgcn_wmma_f32_16x16x32_bf16(
                        false, af[i].v, false, bf[j].v, (short)0, acc[i][j],
                        false, false);
        }
        __syncthreads();
    }

    // Epilogue
#pragma unroll
    for (int i = 0; i < 2; ++i) {
#pragma unroll
        for (int j = 0; j < 4; ++j) {
            int gc = n0 + wc * 64 + j * 16 + ln;
            float bv = bias[gc];
#pragma unroll
            for (int r = 0; r < 8; ++r) {
                int gr = m0 + wr * 32 + i * 16 + r + hlf * 8;
                float v = acc[i][j][r] + bv;
                if (EPI == 0) {
                    ((unsigned short*)outp)[(size_t)gr * N + gc] = f2bf(v);
                } else if (EPI == 1) {
                    int bi = gr / LkV, t = gr % LkV;
                    int hh = gc >> 6, d = gc & 63;
                    ((unsigned short*)outp)[(((size_t)(bi * NHD + hh) * DHD) + d) * LkV + t] = f2bf(v);
                } else if (EPI == 2) {
                    ((float*)outp)[(size_t)gr * N + gc] = v + res[(size_t)gr * N + gc];
                } else { // 3
                    ((unsigned short*)outp)[(size_t)gr * N + gc] = f2bf(fmaxf(v, 0.0f));
                }
            }
        }
    }
}

// ---------------------------------------------------------------------------
// Streaming sigmoid attention.
// Grid: (Lq/128, NH, B). Block: 256 threads = 8 waves; wave w owns 16 q rows.
// Per 32-wide k chunk: S = Q*K^T (4 WMMA/wave), sigmoid (v_exp + v_rcp, both
// TRANS ops that co-execute with WMMA), nontemporal-stream att_map, re-layout
// P through a per-wave LDS tile, O += P*V^T (4 WMMA/wave).
// ---------------------------------------------------------------------------
__global__ __launch_bounds__(256) void attn_sigmoid_kernel(
    const unsigned short* __restrict__ Qb,   // [B*Lq][H] bf16
    const unsigned short* __restrict__ Kb,   // [B*Lk][H] bf16
    const unsigned short* __restrict__ Vtb,  // [B][NH][DH][Lk] bf16
    float* __restrict__ attMap,              // [B][NH][Lq][Lk] f32
    unsigned short* __restrict__ atted)      // [B*Lq][H] bf16
{
    __shared__ __align__(16) unsigned short Qs[128 * 72];
    __shared__ __align__(16) unsigned short Ks[32 * 72];
    __shared__ __align__(16) unsigned short Vs[64 * 40];
    __shared__ __align__(16) unsigned short Ps[8][16 * 40];

    const int tid  = threadIdx.x;
    const int lane = tid & 31;
    const int wid  = tid >> 5;
    const int hlf  = lane >> 4;
    const int ln   = lane & 15;
    const int h    = blockIdx.y;
    const int b    = blockIdx.z;
    const int qBase = blockIdx.x * 128;

    // Load resident Q tile [128][64]
    {
        int row = tid >> 1;
        int cs  = (tid & 1) * 32;
        const uint4* gp = (const uint4*)(Qb + (size_t)(b * LQ + qBase + row) * HH + h * DHD + cs);
        uint4* sp = (uint4*)&Qs[row * 72 + cs];
        sp[0] = gp[0]; sp[1] = gp[1]; sp[2] = gp[2]; sp[3] = gp[3];
    }

    f32x8 accO[4];
#pragma unroll
    for (int j = 0; j < 4; ++j) accO[j] = f32x8{};

    const size_t attRowBase = (size_t)(b * NHD + h) * LQ;
    const unsigned short* gKrow = Kb  + (size_t)(b * LK + (tid >> 3)) * HH + h * DHD + (tid & 7) * 8;
    const unsigned short* gVrow = Vtb + ((size_t)(b * NHD + h) * DHD + (tid >> 2)) * LK + (tid & 3) * 8;

    for (int k0 = 0; k0 < LK; k0 += 32) {
        // prefetch next chunk's K/V rows
        if (k0 + 32 < LK) {
            __builtin_prefetch(gKrow + (size_t)32 * HH, 0, 3);
            __builtin_prefetch(gVrow + 32, 0, 3);
        }
        {   // K chunk: [32 tokens][64]
            int row = tid >> 3, cs = (tid & 7) * 8;
            *(uint4*)&Ks[row * 72 + cs] = *(const uint4*)(gKrow);
        }
        {   // V^T chunk: [64 d][32 k]
            int row = tid >> 2, cs = (tid & 3) * 8;
            *(uint4*)&Vs[row * 40 + cs] = *(const uint4*)(gVrow);
        }
        gKrow += (size_t)32 * HH;
        gVrow += 32;
        __syncthreads();

        const int qr = wid * 16;
#pragma unroll
        for (int n = 0; n < 2; ++n) {
            f32x8 s = f32x8{};
#pragma unroll
            for (int kc = 0; kc < 2; ++kc) {
                FragB a, kb;
                a.q[0]  = *(const uint4*)&Qs[(qr + ln) * 72 + kc * 32 + hlf * 8];
                a.q[1]  = *(const uint4*)&Qs[(qr + ln) * 72 + kc * 32 + 16 + hlf * 8];
                kb.q[0] = *(const uint4*)&Ks[(n * 16 + ln) * 72 + kc * 32 + hlf * 8];
                kb.q[1] = *(const uint4*)&Ks[(n * 16 + ln) * 72 + kc * 32 + 16 + hlf * 8];
                s = __builtin_amdgcn_wmma_f32_16x16x32_bf16(
                        false, a.v, false, kb.v, (short)0, s, false, false);
            }
            // sigmoid, emit att_map (streaming NT store), stash bf16 P
#pragma unroll
            for (int r = 0; r < 8; ++r) {
                int q  = qBase + qr + r + hlf * 8;
                int kg = k0 + n * 16 + ln;
                float e  = __builtin_amdgcn_exp2f(s[r] * SIGK);    // v_exp_f32
                float pv = __builtin_amdgcn_rcpf(1.0f + e);        // v_rcp_f32
                __builtin_nontemporal_store(pv, &attMap[(attRowBase + q) * LK + kg]);
                Ps[wid][(r + hlf * 8) * 40 + n * 16 + ln] = f2bf(pv);
            }
        }
        // O += P * V^T  (same-wave LDS RAW; hw DScnt waits keep it ordered)
#pragma unroll
        for (int j = 0; j < 4; ++j) {
            FragB a, vb;
            a.q[0]  = *(const uint4*)&Ps[wid][ln * 40 + hlf * 8];
            a.q[1]  = *(const uint4*)&Ps[wid][ln * 40 + 16 + hlf * 8];
            vb.q[0] = *(const uint4*)&Vs[(j * 16 + ln) * 40 + hlf * 8];
            vb.q[1] = *(const uint4*)&Vs[(j * 16 + ln) * 40 + 16 + hlf * 8];
            accO[j] = __builtin_amdgcn_wmma_f32_16x16x32_bf16(
                          false, a.v, false, vb.v, (short)0, accO[j], false, false);
        }
        __syncthreads();
    }

#pragma unroll
    for (int j = 0; j < 4; ++j)
#pragma unroll
        for (int r = 0; r < 8; ++r) {
            int q   = qBase + wid * 16 + r + hlf * 8;
            int col = h * DHD + j * 16 + ln;
            atted[(size_t)(b * LQ + q) * HH + col] = f2bf(accO[j][r]);
        }
}

// ---------------------------------------------------------------------------
// LayerNorm (reference semantics: g*(x-mean)/(std+eps)+b, std with ddof=1).
// One 256-thread block per row of 1024.
// ---------------------------------------------------------------------------
__global__ __launch_bounds__(256) void layernorm_kernel(
    const float* __restrict__ in,
    const float* __restrict__ g, const float* __restrict__ be,
    float* __restrict__ outF, unsigned short* __restrict__ outB)
{
    __shared__ float s1[256];
    __shared__ float s2[256];
    const int row = blockIdx.x;
    const int t   = threadIdx.x;
    const float* p = in + (size_t)row * HH;
    float4 v = *(const float4*)(p + t * 4);
    s1[t] = v.x + v.y + v.z + v.w;
    s2[t] = v.x * v.x + v.y * v.y + v.z * v.z + v.w * v.w;
    __syncthreads();
    for (int o = 128; o > 0; o >>= 1) {
        if (t < o) { s1[t] += s1[t + o]; s2[t] += s2[t + o]; }
        __syncthreads();
    }
    float mean = s1[0] * (1.0f / HH);
    float var  = (s2[0] - (float)HH * mean * mean) * (1.0f / (HH - 1));
    float rs   = 1.0f / (sqrtf(fmaxf(var, 0.0f)) + 1e-6f);
    float4 gg = *(const float4*)(g  + t * 4);
    float4 bb = *(const float4*)(be + t * 4);
    float4 o;
    o.x = gg.x * (v.x - mean) * rs + bb.x;
    o.y = gg.y * (v.y - mean) * rs + bb.y;
    o.z = gg.z * (v.z - mean) * rs + bb.z;
    o.w = gg.w * (v.w - mean) * rs + bb.w;
    if (outF) *(float4*)(outF + (size_t)row * HH + t * 4) = o;
    if (outB) {
        unsigned short* q = outB + (size_t)row * HH + t * 4;
        q[0] = f2bf(o.x); q[1] = f2bf(o.y); q[2] = f2bf(o.z); q[3] = f2bf(o.w);
    }
}

// ---------------------------------------------------------------------------
// Host orchestration
// ---------------------------------------------------------------------------
extern "C" void kernel_launch(void* const* d_in, const int* in_sizes, int n_in,
                              void* d_out, int out_size, void* d_ws, size_t ws_size,
                              hipStream_t stream)
{
    const float* x   = (const float*)d_in[0];
    const float* y   = (const float*)d_in[1];
    const float* Wq  = (const float*)d_in[2];
    const float* bq  = (const float*)d_in[3];
    const float* Wk  = (const float*)d_in[4];
    const float* bk  = (const float*)d_in[5];
    const float* Wv  = (const float*)d_in[6];
    const float* bv  = (const float*)d_in[7];
    const float* Wm  = (const float*)d_in[8];
    const float* bm  = (const float*)d_in[9];
    const float* W1  = (const float*)d_in[10];
    const float* b1  = (const float*)d_in[11];
    const float* W2  = (const float*)d_in[12];
    const float* b2  = (const float*)d_in[13];
    const float* g1  = (const float*)d_in[14];
    const float* be1 = (const float*)d_in[15];
    const float* g2  = (const float*)d_in[16];
    const float* be2 = (const float*)d_in[17];

    char* ws = (char*)d_ws;
    size_t off = 0;
    auto alloc = [&](size_t bytes) -> void* {
        void* p = ws + off;
        off += (bytes + 255) & ~(size_t)255;
        return p;
    };

    unsigned short* xb     = (unsigned short*)alloc((size_t)MX * HH * 2);
    unsigned short* yb     = (unsigned short*)alloc((size_t)MY * HH * 2);
    unsigned short* Wqt    = (unsigned short*)alloc((size_t)HH * HH * 2);
    unsigned short* Wkt    = (unsigned short*)alloc((size_t)HH * HH * 2);
    unsigned short* Wvt    = (unsigned short*)alloc((size_t)HH * HH * 2);
    unsigned short* Wmt    = (unsigned short*)alloc((size_t)HH * HH * 2);
    unsigned short* W1t    = (unsigned short*)alloc((size_t)FFD * HH * 2);
    unsigned short* W2t    = (unsigned short*)alloc((size_t)HH * FFD * 2);
    unsigned short* Qbuf   = (unsigned short*)alloc((size_t)MX * HH * 2);
    unsigned short* Kbuf   = (unsigned short*)alloc((size_t)MY * HH * 2);
    unsigned short* Vtbuf  = (unsigned short*)alloc((size_t)MY * HH * 2);
    unsigned short* attedb = (unsigned short*)alloc((size_t)MX * HH * 2);
    float*          t1     = (float*)alloc((size_t)MX * HH * 4);
    float*          hf     = (float*)alloc((size_t)MX * HH * 4);
    unsigned short* hb     = (unsigned short*)alloc((size_t)MX * HH * 2);
    unsigned short* ffb    = (unsigned short*)alloc((size_t)MX * FFD * 2);
    float*          t2     = (float*)alloc((size_t)MX * HH * 4);

    float* outF   = (float*)d_out;                       // [4,1024,1024]
    float* attMap = outF + (size_t)MX * HH;              // [4,16,1024,2048]

    // 1) fp32 -> bf16 activations + transposed weights
    cvt_bf16_kernel<<<(MX * HH) / 1024, 256, 0, stream>>>(x, xb, MX * HH);
    cvt_bf16_kernel<<<(MY * HH) / 1024, 256, 0, stream>>>(y, yb, MY * HH);
    transpose_cvt_kernel<<<dim3(HH / 32,  HH / 32),  256, 0, stream>>>(Wq, Wqt, HH, HH);
    transpose_cvt_kernel<<<dim3(HH / 32,  HH / 32),  256, 0, stream>>>(Wk, Wkt, HH, HH);
    transpose_cvt_kernel<<<dim3(HH / 32,  HH / 32),  256, 0, stream>>>(Wv, Wvt, HH, HH);
    transpose_cvt_kernel<<<dim3(HH / 32,  HH / 32),  256, 0, stream>>>(Wm, Wmt, HH, HH);
    transpose_cvt_kernel<<<dim3(FFD / 32, HH / 32),  256, 0, stream>>>(W1, W1t, HH, FFD);
    transpose_cvt_kernel<<<dim3(HH / 32,  FFD / 32), 256, 0, stream>>>(W2, W2t, FFD, HH);

    // 2) QKV projections
    gemm_bf16_kernel<0><<<dim3(HH / 128, MX / 128), 256, 0, stream>>>(
        xb, Wqt, bq, nullptr, Qbuf, MX, HH, HH, 0);
    gemm_bf16_kernel<0><<<dim3(HH / 128, MY / 128), 256, 0, stream>>>(
        yb, Wkt, bk, nullptr, Kbuf, MY, HH, HH, 0);
    gemm_bf16_kernel<1><<<dim3(HH / 128, MY / 128), 256, 0, stream>>>(
        yb, Wvt, bv, nullptr, Vtbuf, MY, HH, HH, LK);

    // 3) sigmoid attention (writes att_map output + bf16 atted)
    attn_sigmoid_kernel<<<dim3(LQ / 128, NHD, BB), 256, 0, stream>>>(
        Qbuf, Kbuf, Vtbuf, attMap, attedb);

    // 4) output projection + residual  -> t1 = x + atted@Wm + bm
    gemm_bf16_kernel<2><<<dim3(HH / 128, MX / 128), 256, 0, stream>>>(
        attedb, Wmt, bm, x, t1, MX, HH, HH, 0);

    // 5) LN1 -> hf (f32 residual) + hb (bf16 for FFN)
    layernorm_kernel<<<MX, 256, 0, stream>>>(t1, g1, be1, hf, hb);

    // 6) FFN
    gemm_bf16_kernel<3><<<dim3(FFD / 128, MX / 128), 256, 0, stream>>>(
        hb, W1t, b1, nullptr, ffb, MX, FFD, HH, 0);
    gemm_bf16_kernel<2><<<dim3(HH / 128, MX / 128), 256, 0, stream>>>(
        ffb, W2t, b2, hf, t2, MX, HH, FFD, 0);

    // 7) LN2 -> final output
    layernorm_kernel<<<MX, 256, 0, stream>>>(t2, g2, be2, outF, nullptr);
}
